// MultiHeadSelfAttention_63831803953613
// MI455X (gfx1250) — compile-verified, matrix operand order checked by reading, not run on hardware
//
#include <hip/hip_runtime.h>
#include <hip/hip_bf16.h>
#include <math.h>

typedef __bf16 bf16;
typedef __attribute__((ext_vector_type(16))) __bf16 v16bf;
typedef __attribute__((ext_vector_type(8)))  __bf16 v8bf;
typedef __attribute__((ext_vector_type(8)))  float  v8f;
typedef __attribute__((ext_vector_type(4)))  float  f32x4;
typedef unsigned int u32;
typedef __attribute__((ext_vector_type(4))) u32 u32x4;
typedef __attribute__((ext_vector_type(8))) u32 u32x8;

#define DEV __device__ __forceinline__

// Async global->LDS via inline asm (the clang builtin uses CUDA LangAS pointer
// params that cannot be spelled from HIP source; asm sidesteps that).
#define USE_ASYNC 1

static constexpr int Bb = 2, T = 2048, D = 1024, H = 16, DH = 64, RANK = 32;
static constexpr int BT = Bb * T;      // 4096
static constexpr int D3 = 3 * D;       // 3072
static constexpr float SCALE = 0.17677669529663687f;  // 1/sqrt(RANK)

DEV int lane_id() { return threadIdx.x & 31; }

// ---- vectorized LDS helpers ------------------------------------------------
DEV void store8(bf16* dst, v8bf v) {            // 16B aligned
  *(u32x4*)dst = __builtin_bit_cast(u32x4, v);
}
DEV v8bf cvt8(const float* s) {                 // contiguous f32 -> bf16 x8
  v8bf v;
#pragma unroll
  for (int i = 0; i < 8; ++i) v[i] = (bf16)s[i];
  return v;
}
DEV v8bf cvt8s(const float* s, int stride) {    // strided f32 -> bf16 x8
  v8bf v;
#pragma unroll
  for (int i = 0; i < 8; ++i) v[i] = (bf16)s[i * stride];
  return v;
}
DEV void cp16(bf16* dst, const bf16* src) {     // 16B LDS<-global copy
  *(u32x4*)dst = *(const u32x4*)src;
}
DEV void async_cp16(bf16* lds, const bf16* g) { // 16B async global->LDS
#if USE_ASYNC
  // Low 32 bits of a generic LDS pointer are the AS3 byte offset (ISA 10.2).
  u32 loff = (u32)(size_t)lds;
  asm volatile("global_load_async_to_lds_b128 %0, %1, off"
               :: "v"(loff), "v"(g)
               : "memory");
#else
  cp16(lds, g);
#endif
}
DEV void wait_async() {
#if USE_ASYNC
  asm volatile("s_wait_asynccnt 0" ::: "memory");
#endif
}

// ---- WMMA fragment load ----------------------------------------------------
// Generic fragment loader: tile rows = lane-major dim (M for A, N for B^T),
// K contiguous within a row. Two ds_load_b128 per lane.
// Lane l: row = l&15, K-base = (l>>4)*8; elems [kb..kb+7] and [kb+16..kb+23].
DEV v16bf frag_ld(const bf16* base, int ld) {
  const int l = lane_id();
  const bf16* r0 = base + (l & 15) * ld + ((l >> 4) * 8);
  u32x4 lo = *(const u32x4*)(r0);
  u32x4 hi = *(const u32x4*)(r0 + 16);
  u32x8 u;
  u[0] = lo[0]; u[1] = lo[1]; u[2] = lo[2]; u[3] = lo[3];
  u[4] = hi[0]; u[5] = hi[1]; u[6] = hi[2]; u[7] = hi[3];
  return __builtin_bit_cast(v16bf, u);
}

DEV v8f wmma_bf16(v16bf a, v16bf b, v8f c) {
  return __builtin_amdgcn_wmma_f32_16x16x32_bf16(false, a, false, b,
                                                 (short)0, c, false, false);
}

// ---------------------------------------------------------------------------
// Generic GEMM: C[M,N] = A[M,K] @ B[K,N], fp32 in/out, bf16 WMMA core.
// Block tile 64x64, K-tile 64 (two WMMA k-steps). B staged transposed in LDS.
// Requires M%64==0, N%64==0, K%64==0 (true for all call sites).
// ---------------------------------------------------------------------------
__global__ __launch_bounds__(256) void gemm_bf16wmma(
    const float* __restrict__ A, const float* __restrict__ Bmat,
    float* __restrict__ C, int M, int N, int K) {
  constexpr int BM = 64, BN = 64, BK = 64;
  constexpr int LA  = BK + 8;  // 72
  constexpr int LBK = BK + 8;  // 72 (BsT rows are N, K contiguous)
  __shared__ alignas(16) bf16 As[BM * LA];
  __shared__ alignas(16) bf16 BsT[BN * LBK];

  const int tid = threadIdx.x;
  const int wave = tid >> 5;
  const int mi = wave & 3;
  const int ni = wave >> 2;
  const int m0 = blockIdx.y * BM;
  const int n0 = blockIdx.x * BN;

  // fill indices
  const int ar = (tid * 16) / BK, ac = (tid * 16) % BK;  // A: row, col(16-mult)
  const int bn = tid & 63, bk = (tid >> 6) * 16;         // B^T: col n, k-base

  v8f c0 = {}, c1 = {};

  for (int k0 = 0; k0 < K; k0 += BK) {
    {  // A tile 64x64 row-major, fp32->bf16, 2x b128 stores per thread
      const float* g = A + (size_t)(m0 + ar) * K + k0 + ac;
      store8(&As[ar * LA + ac], cvt8(g));
      store8(&As[ar * LA + ac + 8], cvt8(g + 8));
    }
    {  // B^T tile: lane reads a column of B (stride N, lane-coalesced)
      const float* g = Bmat + (size_t)(k0 + bk) * N + n0 + bn;
      store8(&BsT[bn * LBK + bk], cvt8s(g, N));
      store8(&BsT[bn * LBK + bk + 8], cvt8s(g + (size_t)8 * N, N));
    }
    __syncthreads();
#pragma unroll
    for (int kk = 0; kk < BK; kk += 32) {
      v16bf a  = frag_ld(&As[(mi * 16) * LA + kk], LA);
      v16bf b0 = frag_ld(&BsT[(ni * 32) * LBK + kk], LBK);
      v16bf b1 = frag_ld(&BsT[(ni * 32 + 16) * LBK + kk], LBK);
      c0 = wmma_bf16(a, b0, c0);
      c1 = wmma_bf16(a, b1, c1);
    }
    __syncthreads();
  }

  const int l = lane_id();
  const int col = l & 15;
  const int mb = (l >> 4) * 8;
#pragma unroll
  for (int r = 0; r < 8; ++r) {
    size_t row = m0 + mi * 16 + mb + r;
    C[row * N + n0 + ni * 32 + col]      = c0[r];
    C[row * N + n0 + ni * 32 + 16 + col] = c1[r];
  }
}

// ---------------------------------------------------------------------------
// Low-rank projections: ql = q @ W_q_lsr[h] (x core/sqrt(R)), kl = k @ W_k_lsr.
// One block = 64 rows of one (b,h); two passes (q then k). W staged transposed.
// ---------------------------------------------------------------------------
__global__ __launch_bounds__(256) void lsr_kernel(
    const float* __restrict__ qkv,
    const float* __restrict__ Wq, const float* __restrict__ Wk,
    const float* __restrict__ core,
    bf16* __restrict__ qlb, bf16* __restrict__ klb) {
  constexpr int LQ  = DH + 8;  // 72
  constexpr int LDHp = DH + 8; // 72 (wtT rows are rank, dh contiguous)
  __shared__ alignas(16) bf16 qs[64 * LQ];
  __shared__ alignas(16) bf16 wtT[RANK * LDHp];

  const int t0 = blockIdx.x * 64;
  const int h  = blockIdx.y;
  const int b  = blockIdx.z;
  const int tid  = threadIdx.x;
  const int wave = tid >> 5;
  const int mi = wave & 3;
  const int ni = wave >> 2;
  const int l = lane_id();
  const int col = l & 15;
  const int mb  = (l >> 4) * 8;

  const int ar = (tid * 16) / DH, ac = (tid * 16) % DH;  // activation fill
  const int wr = tid & 31, wd = (tid >> 5) * 8;          // W^T fill: rank, dh

  for (int pass = 0; pass < 2; ++pass) {
    const int qkoff = pass * D;  // q at 0, k at +D inside fused qkv
    {  // 64x64 activation tile
      const float* g = qkv + (size_t)(b * T + t0 + ar) * D3 + qkoff + h * DH + ac;
      store8(&qs[ar * LQ + ac], cvt8(g));
      store8(&qs[ar * LQ + ac + 8], cvt8(g + 8));
    }
    {  // W^T tile: lane reads a rank-column of W (stride RANK)
      const float* W = (pass == 0) ? Wq : Wk;
      const float* g = W + (size_t)h * DH * RANK + (size_t)wd * RANK + wr;
      store8(&wtT[wr * LDHp + wd], cvt8s(g, RANK));
    }
    __syncthreads();

    v8f acc = {};
#pragma unroll
    for (int kk = 0; kk < DH; kk += 32) {
      v16bf a   = frag_ld(&qs[(mi * 16) * LQ + kk], LQ);
      v16bf bfr = frag_ld(&wtT[(ni * 16) * LDHp + kk], LDHp);
      acc = wmma_bf16(a, bfr, acc);
    }
    const int ncol = ni * 16 + col;
    const float sc = (pass == 0) ? core[h * RANK + ncol] * SCALE : 1.0f;
    bf16* outp = (pass == 0) ? qlb : klb;
#pragma unroll
    for (int r = 0; r < 8; ++r) {
      int row = t0 + mi * 16 + mb + r;
      outp[((size_t)(b * H + h) * T + row) * RANK + ncol] = (bf16)(acc[r] * sc);
    }
    __syncthreads();
  }
}

// ---------------------------------------------------------------------------
// Flash attention: per (b,h), 64 query rows per block, causal kv chunks of 64.
// S = ql @ kl^T, online softmax in LDS, y += P @ V. V staged transposed.
// ---------------------------------------------------------------------------
__global__ __launch_bounds__(256) void attn_kernel(
    const float* __restrict__ qkv,
    const bf16* __restrict__ qlb, const bf16* __restrict__ klb,
    float* __restrict__ ybuf) {
  constexpr int LQL = RANK + 8;  // 40
  constexpr int LKL = RANK + 8;  // 40  (klt rows = kv, rank contiguous: B^T ready)
  constexpr int LVK = 64 + 8;    // 72  (vtT rows = dh, kv contiguous)
  constexpr int LS  = 68;        // f32 scores
  constexpr int LP  = 72;        // bf16 probs
  __shared__ alignas(16) bf16  qlt[64 * LQL];
  __shared__ alignas(16) bf16  klt[64 * LKL];
  __shared__ alignas(16) bf16  vtT[DH * LVK];
  __shared__ alignas(16) float Sb[64 * LS];
  __shared__ alignas(16) bf16  Pb[64 * LP];
  __shared__ float fbuf[64];
  __shared__ float lbuf[64];

  const int t0 = blockIdx.x * 64;
  const int h  = blockIdx.y;
  const int b  = blockIdx.z;
  const int tid  = threadIdx.x;
  const int wave = tid >> 5;
  const int mi = wave & 3;   // 16-row tile
  const int ni = wave >> 2;  // 32-col group (for S cols and y/DH cols)
  const int l = lane_id();
  const int col = l & 15;
  const int mb  = (l >> 4) * 8;

  const int lr = (tid * 8) / RANK, lc = (tid * 8) % RANK;  // ql/kl fill
  const int vd = tid & 63, vk = (tid >> 6) * 16;           // V^T fill: dh, kv

  {  // ql tile (64x32 bf16), async copy, loaded once
    const bf16* g = qlb + ((size_t)(b * H + h) * T + t0 + lr) * RANK + lc;
    async_cp16(&qlt[lr * LQL + lc], g);
  }

  float m_i = -1e30f, l_i = 0.f;  // row stats, valid in threads 0..63
  v8f y0 = {}, y1 = {};
  wait_async();
  __syncthreads();

  for (int j0 = 0; j0 <= t0; j0 += 64) {
    {  // kl chunk 64x32 bf16 (async)
      const bf16* g = klb + ((size_t)(b * H + h) * T + j0 + lr) * RANK + lc;
      async_cp16(&klt[lr * LKL + lc], g);
    }
    {  // V^T chunk: lane reads a dh-column of V (stride D3, lane-coalesced)
      const float* g = qkv + (size_t)(b * T + j0 + vk) * D3 + 2 * D + h * DH + vd;
      store8(&vtT[vd * LVK + vk], cvt8s(g, D3));
      store8(&vtT[vd * LVK + vk + 8], cvt8s(g + (size_t)8 * D3, D3));
    }
    wait_async();
    __syncthreads();

    // S tiles: A = ql rows (K=32), B = kl^T (klt is [kv][rank] = fragment-ready)
    v16bf aq  = frag_ld(&qlt[(mi * 16) * LQL], LQL);
    v16bf bk0 = frag_ld(&klt[(ni * 32) * LKL], LKL);
    v16bf bk1 = frag_ld(&klt[(ni * 32 + 16) * LKL], LKL);
    v8f zero = {};
    v8f s0 = wmma_bf16(aq, bk0, zero);
    v8f s1 = wmma_bf16(aq, bk1, zero);

    // write scores to LDS with causal mask
#pragma unroll
    for (int r = 0; r < 8; ++r) {
      int row = mi * 16 + mb + r;
      int qg  = t0 + row;
      int c0g = j0 + ni * 32 + col;
      Sb[row * LS + ni * 32 + col]      = (c0g <= qg)      ? s0[r] : -1e30f;
      Sb[row * LS + ni * 32 + 16 + col] = (c0g + 16 <= qg) ? s1[r] : -1e30f;
    }
    __syncthreads();

    // online softmax: one thread per query row, vectorized row sweeps
    if (tid < 64) {
      const float* srow = &Sb[tid * LS];
      const f32x4* s4 = (const f32x4*)srow;
      f32x4 m4 = s4[0];
#pragma unroll
      for (int c = 1; c < 16; ++c) {
        f32x4 v = s4[c];
        m4[0] = fmaxf(m4[0], v[0]); m4[1] = fmaxf(m4[1], v[1]);
        m4[2] = fmaxf(m4[2], v[2]); m4[3] = fmaxf(m4[3], v[3]);
      }
      float mx = fmaxf(fmaxf(m4[0], m4[1]), fmaxf(m4[2], m4[3]));
      float newm = fmaxf(m_i, mx);
      float f = __expf(m_i - newm);
      float sum = 0.f;
      bf16* prow = &Pb[tid * LP];
#pragma unroll
      for (int c0 = 0; c0 < 64; c0 += 8) {
        v8bf pv;
#pragma unroll
        for (int i = 0; i < 8; ++i) {
          float p = __expf(srow[c0 + i] - newm);
          pv[i] = (bf16)p;
          sum += p;
        }
        store8(&prow[c0], pv);
      }
      l_i = l_i * f + sum;
      m_i = newm;
      fbuf[tid] = f;
    }
    __syncthreads();

    // rescale running y, then y += P @ V (K = 64 kv -> two WMMA k-steps)
#pragma unroll
    for (int r = 0; r < 8; ++r) {
      float f = fbuf[mi * 16 + mb + r];
      y0[r] *= f;
      y1[r] *= f;
    }
#pragma unroll
    for (int jj = 0; jj < 2; ++jj) {
      v16bf ap  = frag_ld(&Pb[(mi * 16) * LP + jj * 32], LP);
      v16bf bv0 = frag_ld(&vtT[(ni * 32) * LVK + jj * 32], LVK);
      v16bf bv1 = frag_ld(&vtT[(ni * 32 + 16) * LVK + jj * 32], LVK);
      y0 = wmma_bf16(ap, bv0, y0);
      y1 = wmma_bf16(ap, bv1, y1);
    }
    __syncthreads();
  }

  if (tid < 64) lbuf[tid] = l_i;
  __syncthreads();

#pragma unroll
  for (int r = 0; r < 8; ++r) {
    int row = mi * 16 + mb + r;
    float inv = 1.0f / lbuf[row];
    size_t base = (size_t)(b * T + t0 + row) * D + h * DH;
    ybuf[base + ni * 32 + col]      = y0[r] * inv;
    ybuf[base + ni * 32 + 16 + col] = y1[r] * inv;
  }
}

// ---------------------------------------------------------------------------
// Host-side launch. Workspace: qkv f32 48MB | qlb bf16 4MB | klb bf16 4MB |
// ybuf f32 16MB  (total 72MB).
// ---------------------------------------------------------------------------
extern "C" void kernel_launch(void* const* d_in, const int* in_sizes, int n_in,
                              void* d_out, int out_size, void* d_ws, size_t ws_size,
                              hipStream_t stream) {
  (void)in_sizes; (void)n_in; (void)out_size; (void)ws_size;
  const float* x     = (const float*)d_in[0];
  const float* W_qkv = (const float*)d_in[1];
  const float* Wq    = (const float*)d_in[2];
  const float* Wk    = (const float*)d_in[3];
  const float* core  = (const float*)d_in[4];
  const float* W_o   = (const float*)d_in[5];
  float* out = (float*)d_out;

  char* ws = (char*)d_ws;
  float* qkv = (float*)ws;
  size_t off = (size_t)BT * D3 * sizeof(float);
  bf16* qlb = (bf16*)(ws + off); off += (size_t)Bb * H * T * RANK * sizeof(bf16);
  bf16* klb = (bf16*)(ws + off); off += (size_t)Bb * H * T * RANK * sizeof(bf16);
  float* ybuf = (float*)(ws + off);

  gemm_bf16wmma<<<dim3(D3 / 64, BT / 64), 256, 0, stream>>>(x, W_qkv, qkv, BT, D3, D);
  lsr_kernel<<<dim3(T / 64, H, Bb), 256, 0, stream>>>(qkv, Wq, Wk, core, qlb, klb);
  attn_kernel<<<dim3(T / 64, H, Bb), 256, 0, stream>>>(qkv, qlb, klb, ybuf);
  gemm_bf16wmma<<<dim3(D / 64, BT / 64), 256, 0, stream>>>(ybuf, W_o, out, BT, D, D);
}
